// MOEBatched_73126113181995
// MI455X (gfx1250) — compile-verified, matrix operand order checked by reading, not run on hardware
//
#include <hip/hip_runtime.h>
#include <hip/hip_bf16.h>
#include <stdint.h>

// MoE SwiGLU FFN for MI455X (gfx1250), bf16 WMMA with fp32 accumulate.
// C=1024, F=512, E=128, 512 tokens/expert, T=65536.

#define C_IN   1024
#define C_MID  512
#define N_EXP  128
#define TPE    512

typedef __attribute__((ext_vector_type(16))) __bf16       v16bf;
typedef __attribute__((ext_vector_type(8)))  float        v8f;
typedef __attribute__((ext_vector_type(8)))  unsigned int v8u;

__device__ __forceinline__ unsigned int pack2_bf16(float a, float b) {
  unsigned int ua = __builtin_bit_cast(unsigned int, a);
  unsigned int ub = __builtin_bit_cast(unsigned int, b);
  ua += 0x7FFFu + ((ua >> 16) & 1u);  // round-to-nearest-even
  ub += 0x7FFFu + ((ub >> 16) & 1u);
  return (ua >> 16) | (ub & 0xFFFF0000u);
}

__device__ __forceinline__ unsigned short f2bf(float a) {
  unsigned int u = __builtin_bit_cast(unsigned int, a);
  u += 0x7FFFu + ((u >> 16) & 1u);
  return (unsigned short)(u >> 16);
}

// 32 consecutive fp32 (global) -> 16 packed bf16 pairs (LDS)
__device__ __forceinline__ void g2l_cvt32(const float* __restrict__ g,
                                          unsigned int* __restrict__ l) {
#pragma unroll
  for (int j = 0; j < 4; ++j) {
    float4 fa = ((const float4*)g)[2 * j + 0];
    float4 fb = ((const float4*)g)[2 * j + 1];
    uint4 o;
    o.x = pack2_bf16(fa.x, fa.y);
    o.y = pack2_bf16(fa.z, fa.w);
    o.z = pack2_bf16(fb.x, fb.y);
    o.w = pack2_bf16(fb.z, fb.w);
    ((uint4*)l)[j] = o;
  }
}

// Build a 16x32 bf16 A/B fragment for v_wmma_f32_16x16x32_bf16 from an LDS
// tile stored as [row][64 K] bf16 (pairs packed in uint). Caller passes
// pointer at row*32pairs + ks*16 + kh*4; lo half = K kh*8..+7, hi = K 16+kh*8..+7.
__device__ __forceinline__ v16bf frag_ld(const unsigned int* p) {
  uint4 lo = *(const uint4*)(p);
  uint4 hi = *(const uint4*)(p + 8);
  v8u r;
  r[0] = lo.x; r[1] = lo.y; r[2] = lo.z; r[3] = lo.w;
  r[4] = hi.x; r[5] = hi.y; r[6] = hi.z; r[7] = hi.w;
  return __builtin_bit_cast(v16bf, r);
}

#define WMMA_BF16(A, B, Cc) \
  __builtin_amdgcn_wmma_f32_16x16x32_bf16(false, (A), false, (B), (short)0, (Cc), false, false)

// ---------------------------------------------------------------------------
// Kernel 1: h[e,t,f] = silu(x_g . w1^T + b1) * (x_g . w3^T + b3), bf16 out.
// Block = 128 tokens x 128 F, K over C_IN in chunks of 64. 256 threads.
// ---------------------------------------------------------------------------
__global__ __launch_bounds__(256) void moe_gateup_swiglu(
    const float* __restrict__ x, const int* __restrict__ idxs,
    const float* __restrict__ w1w, const float* __restrict__ w1b,
    const float* __restrict__ w3w, const float* __restrict__ w3b,
    unsigned short* __restrict__ hws) {
  __shared__ unsigned int smem[128 * 32 * 3];           // 48 KB
  unsigned int* sA  = smem;                             // X tile, bf16 pairs
  unsigned int* sB1 = smem + 4096;                      // W1 tile
  unsigned int* sB3 = smem + 8192;                      // W3 tile
  unsigned short* stage = (unsigned short*)smem;        // 128x128 bf16 (32 KB overlay)

  const int tid = threadIdx.x;
  const int e  = blockIdx.z;
  const int ty = blockIdx.y;   // token tile 0..3
  const int tx = blockIdx.x;   // F tile 0..3

  const int r  = tid >> 1;     // tile row this thread stages, 0..127
  const int hf = tid & 1;      // which half of the 64-wide K chunk
  const int c0 = hf * 32;

  const int tok = idxs[e * TPE + ty * 128 + r];
  const float* aG  = x   + (size_t)tok * C_IN + c0;
  const float* b1G = w1w + ((size_t)e * C_MID + tx * 128 + r) * C_IN + c0;
  const float* b3G = w3w + ((size_t)e * C_MID + tx * 128 + r) * C_IN + c0;

  const int lane = tid & 31;
  const int wave = tid >> 5;
  const int wm = wave & 3;     // 4 M-groups of 32 rows
  const int wn = wave >> 2;    // 2 N-groups of 64 cols
  const int lr = lane & 15;
  const int kh = lane >> 4;

  const v8f vzero = {0.f, 0.f, 0.f, 0.f, 0.f, 0.f, 0.f, 0.f};
  v8f acc1[2][4], acc3[2][4];
#pragma unroll
  for (int mi = 0; mi < 2; ++mi)
#pragma unroll
    for (int ni = 0; ni < 4; ++ni) { acc1[mi][ni] = vzero; acc3[mi][ni] = vzero; }

  for (int k0 = 0; k0 < C_IN; k0 += 64) {
    __builtin_prefetch(aG  + k0 + 64, 0, 1);   // global_prefetch_b8 next chunk
    __builtin_prefetch(b1G + k0 + 64, 0, 1);
    __builtin_prefetch(b3G + k0 + 64, 0, 1);
    g2l_cvt32(aG  + k0, &sA [r * 32 + hf * 16]);
    g2l_cvt32(b1G + k0, &sB1[r * 32 + hf * 16]);
    g2l_cvt32(b3G + k0, &sB3[r * 32 + hf * 16]);
    __syncthreads();
#pragma unroll
    for (int ks = 0; ks < 2; ++ks) {
      v16bf fa[2], f1[4], f3[4];
#pragma unroll
      for (int mi = 0; mi < 2; ++mi)
        fa[mi] = frag_ld(&sA[(wm * 32 + mi * 16 + lr) * 32 + ks * 16 + kh * 4]);
#pragma unroll
      for (int ni = 0; ni < 4; ++ni) {
        f1[ni] = frag_ld(&sB1[(wn * 64 + ni * 16 + lr) * 32 + ks * 16 + kh * 4]);
        f3[ni] = frag_ld(&sB3[(wn * 64 + ni * 16 + lr) * 32 + ks * 16 + kh * 4]);
      }
#pragma unroll
      for (int mi = 0; mi < 2; ++mi)
#pragma unroll
        for (int ni = 0; ni < 4; ++ni) {
          acc1[mi][ni] = WMMA_BF16(fa[mi], f1[ni], acc1[mi][ni]);
          acc3[mi][ni] = WMMA_BF16(fa[mi], f3[ni], acc3[mi][ni]);
        }
    }
    __syncthreads();
  }

  // Fused epilogue: bias + silu(h1)*h3 -> LDS stage (transpose) -> coalesced store.
#pragma unroll
  for (int ni = 0; ni < 4; ++ni) {
    const int fl = wn * 64 + ni * 16 + lr;                 // local F col
    const float bb1 = w1b[e * C_MID + tx * 128 + fl];
    const float bb3 = w3b[e * C_MID + tx * 128 + fl];
#pragma unroll
    for (int mi = 0; mi < 2; ++mi) {
#pragma unroll
      for (int v = 0; v < 8; ++v) {
        const int m = wm * 32 + mi * 16 + kh * 8 + v;
        const float h1 = acc1[mi][ni][v] + bb1;
        const float h3 = acc3[mi][ni][v] + bb3;
        const float hv = (h1 / (1.f + __expf(-h1))) * h3;  // silu(h1)*h3
        stage[m * 128 + fl] = f2bf(hv);
      }
    }
  }
  __syncthreads();
  {
    const uint4* src = (const uint4*)(stage + r * 128 + hf * 64);
    uint4* dst = (uint4*)(hws + ((size_t)(e * TPE + ty * 128 + r)) * C_MID +
                          tx * 128 + hf * 64);
#pragma unroll
    for (int j = 0; j < 8; ++j) dst[j] = src[j];
  }
}

// ---------------------------------------------------------------------------
// Kernel 2: out[idxs[g], c] = h[e,t,:] . w2[e,c,:] + b2[e,c]  (scatter rows)
// Block = 128 tokens x 128 C, K over C_MID in chunks of 64. 256 threads.
// A tile (bf16 h) is fetched with GLOBAL_LOAD_ASYNC_TO_LDS_B128 (ASYNCcnt):
// no VGPR round-trip, no VALU. INST_OFFSET applies to both LDS and global
// addresses, so one LDS-base VGPR + one 64-bit addr pair serves all 4 beats.
// ---------------------------------------------------------------------------
__global__ __launch_bounds__(256) void moe_down_scatter(
    const unsigned short* __restrict__ hws, const int* __restrict__ idxs,
    const float* __restrict__ w2w, const float* __restrict__ w2b,
    float* __restrict__ out) {
  __shared__ unsigned int smem[128 * 128];   // 64 KB pool
  unsigned int* sA = smem;                   // h tile (bf16 pairs, 16 KB)
  unsigned int* sB = smem + 4096;            // W2 tile (bf16 pairs, 16 KB)
  float* stage = (float*)smem;               // 128x128 f32 (64 KB overlay)

  const int tid = threadIdx.x;
  const int e  = blockIdx.z;
  const int ty = blockIdx.y;   // token tile 0..3
  const int tx = blockIdx.x;   // C tile 0..7

  const int r  = tid >> 1;
  const int hf = tid & 1;

  const unsigned short* aG =
      hws + ((size_t)(e * TPE + ty * 128 + r)) * C_MID + hf * 32;
  const float* bG = w2w + ((size_t)e * C_IN + tx * 128 + r) * C_MID + hf * 32;

  // LDS byte offset for this thread's A-tile slot (low 32 bits of the
  // generic shared-aperture address are the LDS offset).
  const unsigned int ldsA = (unsigned int)(uintptr_t)(&sA[r * 32 + hf * 16]);

  const int lane = tid & 31, wave = tid >> 5;
  const int wm = wave & 3, wn = wave >> 2;
  const int lr = lane & 15, kh = lane >> 4;

  const v8f vzero = {0.f, 0.f, 0.f, 0.f, 0.f, 0.f, 0.f, 0.f};
  v8f acc[2][4];
#pragma unroll
  for (int mi = 0; mi < 2; ++mi)
#pragma unroll
    for (int ni = 0; ni < 4; ++ni) acc[mi][ni] = vzero;

  for (int k0 = 0; k0 < C_MID; k0 += 64) {
    __builtin_prefetch(bG + k0 + 64, 0, 1);
    {  // h already bf16: async DMA global -> LDS, 4 x b128 per thread
      const unsigned short* gsrc = aG + k0;
      asm volatile(
          "global_load_async_to_lds_b128 %0, %1, off\n\t"
          "global_load_async_to_lds_b128 %0, %1, off offset:16\n\t"
          "global_load_async_to_lds_b128 %0, %1, off offset:32\n\t"
          "global_load_async_to_lds_b128 %0, %1, off offset:48"
          :: "v"(ldsA), "v"(gsrc) : "memory");
    }
    g2l_cvt32(bG + k0, &sB[r * 32 + hf * 16]);
    // all of this wave's async LDS writes landed before we signal the barrier
    asm volatile("s_wait_asynccnt 0" ::: "memory");
    __syncthreads();
#pragma unroll
    for (int ks = 0; ks < 2; ++ks) {
      v16bf fa[2], fb[4];
#pragma unroll
      for (int mi = 0; mi < 2; ++mi)
        fa[mi] = frag_ld(&sA[(wm * 32 + mi * 16 + lr) * 32 + ks * 16 + kh * 4]);
#pragma unroll
      for (int ni = 0; ni < 4; ++ni)
        fb[ni] = frag_ld(&sB[(wn * 64 + ni * 16 + lr) * 32 + ks * 16 + kh * 4]);
#pragma unroll
      for (int mi = 0; mi < 2; ++mi)
#pragma unroll
        for (int ni = 0; ni < 4; ++ni)
          acc[mi][ni] = WMMA_BF16(fa[mi], fb[ni], acc[mi][ni]);
    }
    __syncthreads();
  }

  // bias -> f32 stage
#pragma unroll
  for (int ni = 0; ni < 4; ++ni) {
    const int cl = wn * 64 + ni * 16 + lr;
    const float bb = w2b[e * C_IN + tx * 128 + cl];
#pragma unroll
    for (int mi = 0; mi < 2; ++mi)
#pragma unroll
      for (int v = 0; v < 8; ++v) {
        const int m = wm * 32 + mi * 16 + kh * 8 + v;
        stage[m * 128 + cl] = acc[mi][ni][v] + bb;
      }
  }
  __syncthreads();
  {  // scatter each token row to its original position (coalesced 128B stores)
    const int dest = idxs[e * TPE + ty * 128 + r];
    const float4* src = (const float4*)(stage + r * 128 + hf * 64);
    float4* dst = (float4*)(out + (size_t)dest * C_IN + tx * 128 + hf * 64);
#pragma unroll
    for (int j = 0; j < 16; ++j) dst[j] = src[j];
  }
}

extern "C" void kernel_launch(void* const* d_in, const int* in_sizes, int n_in,
                              void* d_out, int out_size, void* d_ws, size_t ws_size,
                              hipStream_t stream) {
  const float* x    = (const float*)d_in[0];
  const int*   idxs = (const int*)d_in[1];
  const float* w1w  = (const float*)d_in[2];
  const float* w1b  = (const float*)d_in[3];
  const float* w2w  = (const float*)d_in[4];
  const float* w2b  = (const float*)d_in[5];
  const float* w3w  = (const float*)d_in[6];
  const float* w3b  = (const float*)d_in[7];
  float* out = (float*)d_out;
  unsigned short* hws = (unsigned short*)d_ws;  // 128*512*512 bf16 = 64 MB

  dim3 blk(256);
  moe_gateup_swiglu<<<dim3(C_MID / 128, TPE / 128, N_EXP), blk, 0, stream>>>(
      x, idxs, w1w, w1b, w3w, w3b, hws);
  moe_down_scatter<<<dim3(C_IN / 128, TPE / 128, N_EXP), blk, 0, stream>>>(
      hws, idxs, w2w, w2b, out);
}
